// EntityNLM_76905684402292
// MI455X (gfx1250) — matrix operand order repaired and checked by reading, not compiled, not run on hardware
//
#include <hip/hip_runtime.h>
#include <hip/hip_bf16.h>
#include <math.h>

#define VV 50257
#define DD 300
#define HH 128
#define EE 64
#define BB 4
#define TT 512
#define G4 (4*HH)   // 512

typedef __attribute__((ext_vector_type(16))) _Float16 v16h;
typedef __attribute__((ext_vector_type(8)))  _Float16 v8h;
typedef __attribute__((ext_vector_type(8)))  float    v8f;

__device__ __forceinline__ float sigm(float x) { return 1.0f / (1.0f + __expf(-x)); }

__device__ __forceinline__ void sched_fence() {
#if __has_builtin(__builtin_amdgcn_sched_barrier)
    __builtin_amdgcn_sched_barrier(0);   // nothing may be reordered across this
#endif
}

// ---------------------------------------------------------------------------
// Kernel 1: Xg[bt][j] = bih[j] + bhh[j] + dot(Wih[j,:], embed[tokens[bt],:])
// ---------------------------------------------------------------------------
__global__ __launch_bounds__(256)
void xgate_kernel(const int* __restrict__ tokens, const float* __restrict__ embed,
                  const float* __restrict__ Wih, const float* __restrict__ bih,
                  const float* __restrict__ bhh, float* __restrict__ Xg)
{
    const int bt  = blockIdx.x;            // 0..2047
    const int tid = threadIdx.x;
    __shared__ float xr[DD];
    const int tok = tokens[bt];
    for (int d = tid; d < DD; d += 256) xr[d] = embed[(size_t)tok * DD + d];
    __syncthreads();
    for (int j = tid; j < G4; j += 256) {
        const float* w = Wih + (size_t)j * DD;
        float s = bih[j] + bhh[j];
        #pragma unroll 4
        for (int d = 0; d < DD; ++d) s = fmaf(xr[d], w[d], s);
        Xg[(size_t)bt * G4 + j] = s;
    }
}

// ---------------------------------------------------------------------------
// Kernel 2: sequential scan, one block per batch element (512 threads).
// Emits pred_r, pred_e directly, and Zh = (half)(h + ent_feat) to workspace.
// ---------------------------------------------------------------------------
__global__ __launch_bounds__(512)
void scan_kernel(const int* __restrict__ entity_ids, const int* __restrict__ sent_ids,
                 const float* __restrict__ Whh, const float* __restrict__ Wr,
                 const float* __restrict__ brp, const float* __restrict__ We,
                 const float* __restrict__ Wd,  const float* __restrict__ WTe,
                 const float* __restrict__ WTc, const float* __restrict__ ents0,
                 const float* __restrict__ lamp, const float* __restrict__ Xg,
                 _Float16* __restrict__ Zh, float* __restrict__ out_pr,
                 float* __restrict__ out_pe)
{
    const int b   = blockIdx.x;    // 0..3
    const int tid = threadIdx.x;   // 0..511

    __shared__ float entS[EE * HH];   // 32 KB entity stack
    __shared__ float hS[HH], cS[HH], vS[HH], wS[HH], updS[HH], lastS[HH];
    __shared__ float gS[G4];
    __shared__ float distS[EE];
    __shared__ float scal[2];         // [0]=delta, [1]=1/norm

    const float br  = brp[0];
    const float lam = lamp[0];

    for (int i = tid; i < EE * HH; i += 512) entS[i] = ents0[(size_t)b * EE * HH + i];
    if (tid < HH) { hS[tid] = 0.f; cS[tid] = 0.f; lastS[tid] = ents0[(size_t)b * EE * HH + tid]; }
    if (tid < EE) distS[tid] = 0.f;
    __syncthreads();

    for (int t = 0; t < TT; ++t) {
        const int bt  = b * TT + t;
        const int eid = entity_ids[bt];
        const float sid = (float)sent_ids[bt];

        // ---- A: gates = Xg + h @ Whh^T ----
        {
            const float* wr = Whh + (size_t)tid * HH;
            float s = Xg[(size_t)bt * G4 + tid];
            #pragma unroll 8
            for (int k = 0; k < HH; ++k) s = fmaf(hS[k], wr[k], s);
            gS[tid] = s;
        }
        __syncthreads();

        // ---- B: LSTM cell update (gate order i,f,g,o) ----
        if (tid < HH) {
            float ig = sigm(gS[tid]);
            float fg = sigm(gS[HH + tid]);
            float gg = tanhf(gS[2 * HH + tid]);
            float og = sigm(gS[3 * HH + tid]);
            float cn = fg * cS[tid] + ig * gg;
            cS[tid] = cn;
            hS[tid] = og * tanhf(cn);
        }
        __syncthreads();

        // ---- C: v = We@h , w = Wd@h ----
        if (tid < HH) {
            const float* r = We + (size_t)tid * HH;
            float s = 0.f;
            #pragma unroll 8
            for (int k = 0; k < HH; ++k) s = fmaf(r[k], hS[k], s);
            vS[tid] = s;
        } else if (tid < 2 * HH) {
            const int i = tid - HH;
            const float* r = Wd + (size_t)i * HH;
            float s = 0.f;
            #pragma unroll 8
            for (int k = 0; k < HH; ++k) s = fmaf(r[k], hS[k], s);
            wS[i] = s;
        }
        __syncthreads();

        // ---- D: pred_e (threads 0..63), pred_r (wave 2), delta (wave 3) ----
        if (tid < EE) {
            const float* e = entS + (size_t)tid * HH;
            float s = 0.f;
            #pragma unroll 8
            for (int k = 0; k < HH; ++k) s = fmaf(e[k], vS[k], s);
            out_pe[(size_t)bt * EE + tid] = s + __expf((distS[tid] - sid) * lam);
        } else if (tid < 96) {            // wave 2: pred_r = sigmoid(h.Wr + br)
            const int l = tid - 64;
            float s = 0.f;
            #pragma unroll
            for (int j = 0; j < 4; ++j) s = fmaf(hS[l + 32 * j], Wr[l + 32 * j], s);
            #pragma unroll
            for (int off = 16; off > 0; off >>= 1) s += __shfl_xor(s, off, 32);
            if (l == 0) out_pr[bt] = sigm(s + br);
        } else if (tid < 128) {           // wave 3: delta = sigmoid(ent . w)
            const int l = tid - 96;
            const float* e = entS + (size_t)eid * HH;
            float s = 0.f;
            #pragma unroll
            for (int j = 0; j < 4; ++j) s = fmaf(e[l + 32 * j], wS[l + 32 * j], s);
            #pragma unroll
            for (int off = 16; off > 0; off >>= 1) s += __shfl_xor(s, off, 32);
            if (l == 0) scal[0] = sigm(s);
        }
        __syncthreads();

        // ---- E: blended update ----
        if (tid < HH) {
            const float d = scal[0];
            updS[tid] = d * entS[(size_t)eid * HH + tid] + (1.f - d) * hS[tid];
        }
        __syncthreads();

        // ---- F: norm (wave 0) ----
        if (tid < 32) {
            float s = 0.f;
            #pragma unroll
            for (int j = 0; j < 4; ++j) { float u = updS[tid + 32 * j]; s = fmaf(u, u, s); }
            #pragma unroll
            for (int off = 16; off > 0; off >>= 1) s += __shfl_xor(s, off, 32);
            if (tid == 0) scal[1] = 1.0f / sqrtf(s);
        }
        __syncthreads();

        // ---- G: normalize, scatter into ents/dists ----
        if (tid < HH) {
            float un = updS[tid] * scal[1];
            updS[tid] = un;
            entS[(size_t)eid * HH + tid] = un;
        }
        if (tid == 0) distS[eid] = sid;
        __syncthreads();

        // ---- H: ent_feat + z ----
        if (tid < HH) {
            float s = 0.f;
            if (eid > 0) {
                const float* r = WTe + (size_t)tid * HH;
                #pragma unroll 8
                for (int k = 0; k < HH; ++k) s = fmaf(r[k], updS[k], s);
            } else {
                const float* r = WTc + (size_t)tid * HH;
                #pragma unroll 8
                for (int k = 0; k < HH; ++k) s = fmaf(r[k], lastS[k], s);
            }
            Zh[(size_t)bt * HH + tid] = (_Float16)(hS[tid] + s);
        }
        __syncthreads();
        if (tid < HH) lastS[tid] = updS[tid];
        __syncthreads();
    }
}

// ---------------------------------------------------------------------------
// Kernel 3: f32 -> f16 convert (for Wx)
// ---------------------------------------------------------------------------
__global__ __launch_bounds__(256)
void cvt_half_kernel(const float* __restrict__ x, _Float16* __restrict__ y, int n)
{
    int i = blockIdx.x * blockDim.x + threadIdx.x;
    const int stride = gridDim.x * blockDim.x;
    for (; i < n; i += stride) y[i] = (_Float16)x[i];
}

// ---------------------------------------------------------------------------
// Kernel 4: logits[2048][V] = Zh(2048x128,f16) @ Wxh^T + bx using
// v_wmma_f32_16x16x32_f16. Each wave owns one 16-row M-tile (A held in 32
// VGPRs across the whole N loop), loops over 8 consecutive N-tiles.
// B[k][n] = Wxh[n][k] => K is the contiguous axis, so each lane's B fragment
// is 32 contiguous bytes: one v16h load = 2 clause-paired b128 loads.
// sched_barrier forces: [issue all 8 B loads + bias] | [4 WMMAs] so the
// waitcnt inserter can use partial waits and overlap WMMA with in-flight loads.
// ---------------------------------------------------------------------------
__global__ __launch_bounds__(256)
void logits_wmma_kernel(const _Float16* __restrict__ Zh, const _Float16* __restrict__ Wxh,
                        const float* __restrict__ bx, float* __restrict__ out)
{
    const int lane = threadIdx.x & 31;
    const int wave = threadIdx.x >> 5;                 // 0..7
    const int m0   = (blockIdx.y * 8 + wave) * 16;     // 0..2032
    const int hi   = lane >> 4;                        // lane-half select
    const int l15  = lane & 15;

    // A fragments: 4 K-chunks of 32.  Layout matching B (16-bit A 16x32):
    // lanes 0-15: K 0..15; lanes 16-31: K 16..31
    v16h a[4];
    const _Float16* zrow = Zh + (size_t)(m0 + l15) * HH;
    #pragma unroll
    for (int kc = 0; kc < 4; ++kc) {
        const int kb = kc * 32 + hi * 16;
        v8h lo  = *(const v8h*)(zrow + kb);
        v8h hi8 = *(const v8h*)(zrow + kb + 8);
        v16h av;
        #pragma unroll
        for (int i = 0; i < 8; ++i) { av[i] = lo[i]; av[8 + i] = hi8[i]; }
        a[kc] = av;
    }

    #pragma unroll 1
    for (int it = 0; it < 8; ++it) {
        const int n0 = (blockIdx.x * 8 + it) * 16;
        if (n0 >= VV) break;
        const int col  = n0 + l15;
        const int colc = col < VV ? col : VV - 1;
        const _Float16* wrow = Wxh + (size_t)colc * HH;

        // (a) issue ALL B loads for this tile (8x global_load_b128) + bias load
        v16h bv[4];
        #pragma unroll
        for (int kc = 0; kc < 4; ++kc) {
            // B layout (32x16, f16): lanes 0-15: N=lane, K 0..15; lanes 16-31: K 16..31
            bv[kc] = *(const v16h*)(wrow + kc * 32 + hi * 16);
        }
        const float bias = bx[colc];

        // hard fence: loads above may not be sunk below; B frags stay live
        sched_fence();

        // (b) 4 back-to-back WMMAs (partial loadcnt waits overlap with loads)
        v8f acc = {};
        #pragma unroll
        for (int kc = 0; kc < 4; ++kc) {
            acc = __builtin_amdgcn_wmma_f32_16x16x32_f16(
                false, a[kc], false, bv[kc], (short)0, acc, false, false);
        }

        if (col < VV) {
            #pragma unroll
            for (int r = 0; r < 8; ++r) {
                const int m = m0 + r + hi * 8;     // C/D layout: vgpr r -> row r / r+8
                out[(size_t)m * VV + col] = acc[r] + bias;
            }
        }
    }
}

// ---------------------------------------------------------------------------
extern "C" void kernel_launch(void* const* d_in, const int* in_sizes, int n_in,
                              void* d_out, int out_size, void* d_ws, size_t ws_size,
                              hipStream_t stream)
{
    const int*   tokens     = (const int*)  d_in[0];
    const int*   entity_ids = (const int*)  d_in[1];
    const int*   sent_ids   = (const int*)  d_in[2];
    const float* embed      = (const float*)d_in[3];
    const float* Wih        = (const float*)d_in[4];
    const float* Whh        = (const float*)d_in[5];
    const float* bih        = (const float*)d_in[6];
    const float* bhh        = (const float*)d_in[7];
    const float* Wr         = (const float*)d_in[8];
    const float* br         = (const float*)d_in[9];
    const float* We         = (const float*)d_in[10];
    const float* Wd         = (const float*)d_in[11];
    const float* WTe        = (const float*)d_in[12];
    const float* WTc        = (const float*)d_in[13];
    const float* Wx         = (const float*)d_in[14];
    const float* bx         = (const float*)d_in[15];
    const float* ents0      = (const float*)d_in[16];
    const float* lam        = (const float*)d_in[17];

    float* out_logits = (float*)d_out;                                  // (B,T,V)
    float* out_pr     = out_logits + (size_t)BB * TT * VV;              // (B,T)
    float* out_pe     = out_pr + (size_t)BB * TT;                       // (B,T,E)

    // workspace layout
    float*    Xg  = (float*)d_ws;                                       // 2048*512 f32 = 4 MiB
    _Float16* Zh  = (_Float16*)((char*)d_ws + (size_t)4 * 1024 * 1024); // 2048*128 f16 = 512 KiB
    _Float16* Wxh = (_Float16*)((char*)d_ws + (size_t)4 * 1024 * 1024 + 512 * 1024); // V*128 f16

    // 1) parallel input-gate precompute
    xgate_kernel<<<BB * TT, 256, 0, stream>>>(tokens, embed, Wih, bih, bhh, Xg);

    // 3) Wx -> f16 (independent; same stream keeps ordering before GEMM)
    cvt_half_kernel<<<6284, 256, 0, stream>>>(Wx, Wxh, VV * HH);

    // 2) sequential recurrence
    scan_kernel<<<BB, 512, 0, stream>>>(entity_ids, sent_ids, Whh, Wr, br, We, Wd,
                                        WTe, WTc, ents0, lam, Xg, Zh, out_pr, out_pe);

    // 4) big WMMA GEMM: logits = Zh @ Wxh^T + bx
    dim3 grid((3142 + 7) / 8, (BB * TT) / 128);   // (393, 16)
    logits_wmma_kernel<<<grid, 256, 0, stream>>>(Zh, Wxh, bx, out_logits);
}
